// Dynamic_LSQConv2d_77180562309712
// MI455X (gfx1250) — 8-bit matrix operand order repaired and checked by reading, compiled, not run on hardware
//
#include <hip/hip_runtime.h>
#include <hip/hip_bf16.h>

typedef int v8i __attribute__((ext_vector_type(8)));
typedef unsigned int uint;

#define BATCH  64
#define CIN    128
#define COUT   128
#define HH     56
#define WWID   56
#define HWPIX  3136
#define KSEL   4
#define HIDDEN 33

#define XQ_OFF   ((size_t)0)
#define WQ_OFF   (XQ_OFF + (size_t)BATCH * HWPIX * CIN)
#define POOL_OFF (WQ_OFF + (size_t)KSEL * 9 * COUT * CIN)
#define SEL_OFF  (POOL_OFF + (size_t)BATCH * CIN * 4)
#define AA_OFF   (SEL_OFF + (size_t)BATCH * 4)
#define RQ_OFF   (AA_OFF  + (size_t)BATCH * 4)
#define SC_OFF   (RQ_OFF  + (size_t)BATCH * 4)
#define OUT_MAIN ((size_t)BATCH * COUT * HWPIX)

__global__ __launch_bounds__(256) void pool_kernel(const float* __restrict__ x,
                                                   float* __restrict__ pooled) {
  __shared__ float s[256];
  const int bc = blockIdx.x;
  const float* p = x + (size_t)bc * HWPIX;
  float acc = 0.f;
  for (int i = threadIdx.x; i < HWPIX; i += 256) acc += p[i];
  s[threadIdx.x] = acc;
  __syncthreads();
  for (int st = 128; st > 0; st >>= 1) {
    if ((int)threadIdx.x < st) s[threadIdx.x] += s[threadIdx.x + st];
    __syncthreads();
  }
  if (threadIdx.x == 0) pooled[bc] = s[0] * (1.0f / (float)HWPIX);
}

__global__ __launch_bounds__(64) void mlp_kernel(const float* __restrict__ pooled,
                                                 const float* __restrict__ fc1,
                                                 const float* __restrict__ fc2,
                                                 const float* __restrict__ fc2b,
                                                 const float* __restrict__ alpha_w,
                                                 const float* __restrict__ alpha_a,
                                                 float* __restrict__ raw_out,
                                                 int* __restrict__ sel,
                                                 float* __restrict__ a_arr,
                                                 float* __restrict__ rqp_arr,
                                                 float* __restrict__ sc_arr) {
  __shared__ float ps[CIN];
  __shared__ float hs[HIDDEN];
  __shared__ float rs[KSEL];
  const int b = blockIdx.x, t = threadIdx.x;
  ps[t]      = pooled[b * CIN + t];
  ps[t + 64] = pooled[b * CIN + t + 64];
  __syncthreads();
  if (t < HIDDEN) {
    float h = 0.f;
    for (int c = 0; c < CIN; ++c) h += ps[c] * fc1[t * CIN + c];
    hs[t] = fmaxf(h, 0.f);
  }
  __syncthreads();
  if (t < KSEL) {
    float r = fc2b[t];
    for (int j = 0; j < HIDDEN; ++j) r += hs[j] * fc2[t * HIDDEN + j];
    rs[t] = r;
    raw_out[b * KSEL + t] = r;
  }
  __syncthreads();
  if (t == 0) {
    int s0 = 0; float best = rs[0];
    for (int i = 1; i < KSEL; ++i) if (rs[i] > best) { best = rs[i]; s0 = i; }
    sel[b]     = s0;
    const float a = alpha_a[s0];
    a_arr[b]   = a;
    rqp_arr[b] = (float)((1 << (s0 + 2)) - 1);
    sc_arr[b]  = a * alpha_w[s0];
  }
}

__global__ __launch_bounds__(128) void wquant_kernel(const float* __restrict__ weight,
                                                     const float* __restrict__ alpha_w,
                                                     signed char* __restrict__ wq) {
  const int bi = blockIdx.x;
  const int k  = bi / (9 * COUT);
  const int p  = (bi / COUT) % 9;
  const int o  = bi % COUT;
  const int c  = threadIdx.x;
  const float aw = alpha_w[k];
  const float qp = (float)((1 << (k + 1)) - 1);
  const float qn = -(float)(1 << (k + 1));
  const float w  = weight[((size_t)((k * COUT + o) * CIN + c)) * 9 + p];
  const float q  = rintf(fminf(fmaxf(w / aw, qn), qp));
  wq[(((size_t)(k * 9 + p) * COUT + o) * CIN) + c] = (signed char)(int)q;
}

__global__ __launch_bounds__(224) void aquant_kernel(const float* __restrict__ x,
                                                     const float* __restrict__ a_arr,
                                                     const float* __restrict__ rqp_arr,
                                                     unsigned char* __restrict__ xq) {
  const int b  = blockIdx.y;
  const int hw = blockIdx.x * 224 + threadIdx.x;
  const float a = a_arr[b], rqp = rqp_arr[b];
  const float* xb = x + (size_t)b * CIN * HWPIX + hw;
  unsigned char* dst = xq + ((size_t)(b * HWPIX + hw)) * CIN;
  for (int cc = 0; cc < CIN; cc += 16) {
    uint u[4];
#pragma unroll
    for (int d = 0; d < 4; ++d) {
      uint v = 0;
#pragma unroll
      for (int j = 0; j < 4; ++j) {
        const int c = cc + d * 4 + j;
        const float q = rintf(fminf(fmaxf(xb[(size_t)c * HWPIX] / a, 0.f), rqp));
        v |= ((uint)(int)q & 0xFFu) << (8 * j);
      }
      u[d] = v;
    }
    *(uint4*)(dst + cc) = make_uint4(u[0], u[1], u[2], u[3]);
  }
}

__global__ __launch_bounds__(256) void conv_wmma_kernel(const signed char* __restrict__ wq,
                                                        const unsigned char* __restrict__ xq,
                                                        const int* __restrict__ sel,
                                                        const float* __restrict__ sc_arr,
                                                        const float* __restrict__ bias,
                                                        float* __restrict__ out) {
  const int b    = blockIdx.y;
  const int nb   = blockIdx.x * 32;
  const int wave = threadIdx.x >> 5;
  const int lane = threadIdx.x & 31;
  const int mbase = wave * 16;
  const int colx  = lane & 15;
  const int hi    = lane >> 4;

  const int   ks = sel[b];
  const float sc = sc_arr[b];

  v8i acc0 = {0, 0, 0, 0, 0, 0, 0, 0};
  v8i acc1 = {0, 0, 0, 0, 0, 0, 0, 0};

  const int o_row = mbase + colx;
  const int n0 = nb + colx;
  const int n1 = n0 + 16;
  const int h0 = n0 / WWID, w0 = n0 % WWID;
  const int h1 = n1 / WWID, w1 = n1 % WWID;

  const unsigned char* xqb = xq + (size_t)b * HWPIX * CIN;

  for (int p = 0; p < 9; ++p) {
    const int dh = p / 3 - 1, dw = p % 3 - 1;
    const int hs0 = h0 + dh, ws0 = w0 + dw;
    const int hs1 = h1 + dh, ws1 = w1 + dw;
    const bool v0 = ((uint)hs0 < (uint)HH) && ((uint)ws0 < (uint)WWID);
    const bool v1 = ((uint)hs1 < (uint)HH) && ((uint)ws1 < (uint)WWID);
    const unsigned char* px0 = xqb + (size_t)(hs0 * WWID + ws0) * CIN;
    const unsigned char* px1 = xqb + (size_t)(hs1 * WWID + ws1) * CIN;
    const signed char*  wrow = wq + (((size_t)(ks * 9 + p) * COUT + o_row) * CIN);

#pragma unroll
    for (int cc = 0; cc < 2; ++cc) {
      const int cb = cc * 64;
      v8i A;
#pragma unroll
      for (int j = 0; j < 4; ++j) {
        const uint2 d = *(const uint2*)(wrow + cb + (j >> 1) * 32 + hi * 16 + (j & 1) * 8);
        A[2 * j]     = (int)d.x;
        A[2 * j + 1] = (int)d.y;
      }
      uint4 lo0 = make_uint4(0, 0, 0, 0), hc0 = make_uint4(0, 0, 0, 0);
      uint4 lo1 = make_uint4(0, 0, 0, 0), hc1 = make_uint4(0, 0, 0, 0);
      if (v0) {
        lo0 = *(const uint4*)(px0 + cb + hi * 16);
        hc0 = *(const uint4*)(px0 + cb + hi * 16 + 32);
      }
      if (v1) {
        lo1 = *(const uint4*)(px1 + cb + hi * 16);
        hc1 = *(const uint4*)(px1 + cb + hi * 16 + 32);
      }
      v8i B0, B1;
      B0[0] = (int)lo0.x; B0[1] = (int)lo0.y; B0[2] = (int)lo0.z; B0[3] = (int)lo0.w;
      B0[4] = (int)hc0.x; B0[5] = (int)hc0.y; B0[6] = (int)hc0.z; B0[7] = (int)hc0.w;
      B1[0] = (int)lo1.x; B1[1] = (int)lo1.y; B1[2] = (int)lo1.z; B1[3] = (int)lo1.w;
      B1[4] = (int)hc1.x; B1[5] = (int)hc1.y; B1[6] = (int)hc1.z; B1[7] = (int)hc1.w;

      acc0 = __builtin_amdgcn_wmma_i32_16x16x64_iu8(true, A, false, B0, acc0, false, false);
      acc1 = __builtin_amdgcn_wmma_i32_16x16x64_iu8(true, A, false, B1, acc1, false, false);
    }
  }

  const float* biasr = bias + ks * COUT;
  float* outb = out + (size_t)b * COUT * HWPIX;
#pragma unroll
  for (int r = 0; r < 8; ++r) {
    const int m  = mbase + r + hi * 8;
    const float bv = biasr[m];
    float* orow = outb + (size_t)m * HWPIX;
    orow[nb + colx]      = (float)acc0[r] * sc + bv;
    orow[nb + colx + 16] = (float)acc1[r] * sc + bv;
  }
}

extern "C" void kernel_launch(void* const* d_in, const int* in_sizes, int n_in,
                              void* d_out, int out_size, void* d_ws, size_t ws_size,
                              hipStream_t stream) {
  const float* x       = (const float*)d_in[0];
  const float* fc1_w   = (const float*)d_in[1];
  const float* fc2_w   = (const float*)d_in[2];
  const float* fc2_b   = (const float*)d_in[3];
  const float* alpha_w = (const float*)d_in[4];
  const float* alpha_a = (const float*)d_in[5];
  const float* weight  = (const float*)d_in[6];
  const float* bias    = (const float*)d_in[7];
  float* out = (float*)d_out;

  char* ws = (char*)d_ws;
  unsigned char* xq = (unsigned char*)(ws + XQ_OFF);
  signed char*   wq = (signed char*)(ws + WQ_OFF);
  float* pooled = (float*)(ws + POOL_OFF);
  int*   sel    = (int*)(ws + SEL_OFF);
  float* a_arr  = (float*)(ws + AA_OFF);
  float* rq_arr = (float*)(ws + RQ_OFF);
  float* sc_arr = (float*)(ws + SC_OFF);

  pool_kernel<<<BATCH * CIN, 256, 0, stream>>>(x, pooled);
  mlp_kernel<<<BATCH, 64, 0, stream>>>(pooled, fc1_w, fc2_w, fc2_b, alpha_w, alpha_a,
                                       out + OUT_MAIN, sel, a_arr, rq_arr, sc_arr);
  wquant_kernel<<<KSEL * 9 * COUT, 128, 0, stream>>>(weight, alpha_w, wq);
  aquant_kernel<<<dim3(14, BATCH), 224, 0, stream>>>(x, a_arr, rq_arr, xq);
  conv_wmma_kernel<<<dim3(HWPIX / 32, BATCH), 256, 0, stream>>>(wq, xq, sel, sc_arr, bias, out);
}
